// Attention_25108378813046
// MI455X (gfx1250) — compile-verified, matrix operand order checked by reading, not run on hardware
//
#include <hip/hip_runtime.h>

typedef __bf16 bf16;
typedef __attribute__((ext_vector_type(16))) __bf16 v16bf;
typedef __attribute__((ext_vector_type(8)))  float  v8f;

// ---------------- CDNA5 async global->LDS copies via inline asm -------------
// global_load_async_to_lds_b128: VDST = LDS byte address (low 32 bits of the
// generic pointer, per the ISA aperture mapping), VADDR = 64-bit global addr.
// Tracked by ASYNCcnt; completion enforced with s_wait_asynccnt.

#if defined(__HIP_DEVICE_COMPILE__) && defined(__gfx1250__)
#define ASYNC_LDS 1
#else
#define ASYNC_LDS 0
#endif

__device__ __forceinline__ void async_copy16(const void* g, void* l) {
#if ASYNC_LDS
  unsigned lds32 = (unsigned)(size_t)l;           // low 32 bits of the pointer give the LDS address
  unsigned long long ga = (unsigned long long)(size_t)g;
  asm volatile("global_load_async_to_lds_b128 %0, %1, off"
               :: "v"(lds32), "v"(ga) : "memory");
#else
  (void)g; (void)l;
#endif
}

__device__ __forceinline__ void async_wait() {
#if ASYNC_LDS
  asm volatile("s_wait_asynccnt 0" ::: "memory");
#endif
}

// ---------------- WMMA helpers (16x16x32 bf16, f32 accum) -------------------

__device__ __forceinline__ v8f wmma_bf16(v16bf a, v16bf b, v8f c) {
  return __builtin_amdgcn_wmma_f32_16x16x32_bf16(false, a, false, b, (short)0, c,
                                                 false, false);
}

__device__ __forceinline__ v8f zero8() {
  v8f z;
#pragma unroll
  for (int i = 0; i < 8; ++i) z[i] = 0.0f;
  return z;
}

// Fragment loader following the ISA 16-bit A/B VGPR layouts.
// idx = lane&15 (row M for A, column N for B); half = lane>>4.
__device__ __forceinline__ v16bf load_frag(const bf16* base, int idx, int half,
                                           int rowStride, int kStride, int kBase) {
  v16bf f;
  const bf16* p = base + idx * rowStride;
#pragma unroll
  for (int v = 0; v < 8; ++v) {
    int K = ((v < 4) ? (2 * v) : (16 + 2 * (v - 4))) + 8 * half + kBase;
    f[2 * v]     = p[K * kStride];
    f[2 * v + 1] = p[(K + 1) * kStride];
  }
  return f;
}

// ---------------- fp32 -> bf16 convert --------------------------------------

__global__ __launch_bounds__(256) void cvt_bf16(const float* __restrict__ in,
                                                bf16* __restrict__ out, int n) {
  int i = blockIdx.x * blockDim.x + threadIdx.x;
  int stride = gridDim.x * blockDim.x;
  for (; i < n; i += stride) out[i] = (bf16)in[i];
}

// ---------------- Unified bf16 GEMM, C[m,n] = sum_k A[m,k]*W[n,k] -----------
// Block tile 128 x BN, BK=32, double-buffered LDS, 8 waves (2M x 4N),
// wave tile 64 x (WNT*16).  MODE 0: QKV epilogue (bias, Q*scale, scatter to
// [3][B,H,N,Dh] bf16).  MODE 1: projection epilogue (bias, fp32 out).

template <int BN, int WNT, int MODE>
__global__ __launch_bounds__(256) void gemm_bf16_kernel(
    const bf16* __restrict__ X, const bf16* __restrict__ W,
    const float* __restrict__ bias, void* __restrict__ outp) {
  constexpr int BU = BN / 64;                 // uint4 passes for the B tile
  __shared__ __align__(16) bf16 sA[2][128 * 32];
  __shared__ __align__(16) bf16 sB[2][BN * 32];
  const int tid = threadIdx.x;
  const int w = tid >> 5, lane = tid & 31, half = lane >> 4, l15 = lane & 15;
  const int bm = blockIdx.y * 128, bn = blockIdx.x * BN;
  const int wm = (w & 1) * 64, wn = (w >> 1) * (WNT * 16);

  v8f acc[4][WNT];
#pragma unroll
  for (int i = 0; i < 4; ++i)
#pragma unroll
    for (int j = 0; j < WNT; ++j) acc[i][j] = zero8();

  // ---- prologue: stage K-tile 0 into buffer 0
#if ASYNC_LDS
#pragma unroll
  for (int j = 0; j < 2; ++j) {
    int u = j * 256 + tid, row = u >> 2, seg = u & 3;
    async_copy16(X + (size_t)(bm + row) * 1024 + seg * 8, &sA[0][row * 32 + seg * 8]);
  }
#pragma unroll
  for (int j = 0; j < BU; ++j) {
    int u = j * 256 + tid, row = u >> 2, seg = u & 3;
    async_copy16(W + (size_t)(bn + row) * 1024 + seg * 8, &sB[0][row * 32 + seg * 8]);
  }
  async_wait();
#else
  {
    uint4 ar[2], br[BU];
#pragma unroll
    for (int j = 0; j < 2; ++j) {
      int u = j * 256 + tid, row = u >> 2, seg = u & 3;
      ar[j] = *(const uint4*)(X + (size_t)(bm + row) * 1024 + seg * 8);
    }
#pragma unroll
    for (int j = 0; j < BU; ++j) {
      int u = j * 256 + tid, row = u >> 2, seg = u & 3;
      br[j] = *(const uint4*)(W + (size_t)(bn + row) * 1024 + seg * 8);
    }
#pragma unroll
    for (int j = 0; j < 2; ++j) ((uint4*)sA[0])[j * 256 + tid] = ar[j];
#pragma unroll
    for (int j = 0; j < BU; ++j) ((uint4*)sB[0])[j * 256 + tid] = br[j];
  }
#endif
  __syncthreads();

  int buf = 0;
  for (int k0 = 0; k0 < 1024; k0 += 32) {
    const bool hasNext = (k0 + 32) < 1024;
#if ASYNC_LDS
    if (hasNext) {                            // overlap next tile with compute
#pragma unroll
      for (int j = 0; j < 2; ++j) {
        int u = j * 256 + tid, row = u >> 2, seg = u & 3;
        async_copy16(X + (size_t)(bm + row) * 1024 + k0 + 32 + seg * 8,
                     &sA[buf ^ 1][row * 32 + seg * 8]);
      }
#pragma unroll
      for (int j = 0; j < BU; ++j) {
        int u = j * 256 + tid, row = u >> 2, seg = u & 3;
        async_copy16(W + (size_t)(bn + row) * 1024 + k0 + 32 + seg * 8,
                     &sB[buf ^ 1][row * 32 + seg * 8]);
      }
    }
#else
    uint4 ar[2], br[BU];
    if (hasNext) {
#pragma unroll
      for (int j = 0; j < 2; ++j) {
        int u = j * 256 + tid, row = u >> 2, seg = u & 3;
        ar[j] = *(const uint4*)(X + (size_t)(bm + row) * 1024 + k0 + 32 + seg * 8);
      }
#pragma unroll
      for (int j = 0; j < BU; ++j) {
        int u = j * 256 + tid, row = u >> 2, seg = u & 3;
        br[j] = *(const uint4*)(W + (size_t)(bn + row) * 1024 + k0 + 32 + seg * 8);
      }
    }
#endif
    if (k0 + 64 < 1024) {                     // warm L2 for tile after next
      int row = tid >> 2, seg = tid & 3;
      __builtin_prefetch(X + (size_t)(bm + row) * 1024 + k0 + 64 + seg * 8, 0, 1);
      __builtin_prefetch(W + (size_t)(bn + row) * 1024 + k0 + 64 + seg * 8, 0, 1);
    }

    v16bf afr[4];
#pragma unroll
    for (int i = 0; i < 4; ++i)
      afr[i] = load_frag(&sA[buf][(wm + i * 16) * 32], l15, half, 32, 1, 0);
#pragma unroll
    for (int j = 0; j < WNT; ++j) {
      v16bf bfr = load_frag(&sB[buf][(wn + j * 16) * 32], l15, half, 32, 1, 0);
#pragma unroll
      for (int i = 0; i < 4; ++i) acc[i][j] = wmma_bf16(afr[i], bfr, acc[i][j]);
    }

#if ASYNC_LDS
    async_wait();
#else
    if (hasNext) {
#pragma unroll
      for (int j = 0; j < 2; ++j) ((uint4*)sA[buf ^ 1])[j * 256 + tid] = ar[j];
#pragma unroll
      for (int j = 0; j < BU; ++j) ((uint4*)sB[buf ^ 1])[j * 256 + tid] = br[j];
    }
#endif
    __syncthreads();
    buf ^= 1;
  }

  // ---- epilogue
#pragma unroll
  for (int i = 0; i < 4; ++i)
#pragma unroll
    for (int j = 0; j < WNT; ++j)
#pragma unroll
      for (int r = 0; r < 8; ++r) {
        int m = bm + wm + i * 16 + r + 8 * half;
        int n = bn + wn + j * 16 + l15;
        float v = acc[i][j][r] + bias[n];
        if (MODE == 0) {
          int which = n >> 10;                // 0=q 1=k 2=v
          if (which == 0) v *= 0.125f;        // Dh^-0.5 (Dh=64)
          int c = n & 1023;
          int hh = c >> 6, dd = c & 63;
          int bb = m >> 11, ii = m & 2047;
          ((bf16*)outp)[(size_t)which * 4194304 +
                        ((size_t)((bb * 16 + hh) * 2048 + ii) << 6) + dd] = (bf16)v;
        } else {
          ((float*)outp)[(size_t)m * 1024 + n] = v;
        }
      }
}

// ---------------- Flash attention: one block per (b,h, 128-row Q tile) ------

__global__ __launch_bounds__(256) void attn_fwd(const bf16* __restrict__ Q,
                                                const bf16* __restrict__ K,
                                                const bf16* __restrict__ V,
                                                bf16* __restrict__ O) {
  __shared__ __align__(16) bf16 smem[49152];  // 96 KB
  // layout: [0,8192) K buf0 | [8192,16384) K buf1 | [16384,24576) V buf0 |
  //         [24576,32768) V buf1 | [32768,49152) P (8 waves x 16 x 128), Q@start
  bf16* sP = smem + 32768;
  bf16* sQ = sP;

  const int tid = threadIdx.x;
  const int w = tid >> 5, lane = tid & 31, half = lane >> 4, l15 = lane & 15;
  const int bh = blockIdx.y;                 // b*16 + h
  const int m0 = blockIdx.x * 128;
  const bf16* Qb = Q + (size_t)bh * 2048 * 64;
  const bf16* Kb = K + (size_t)bh * 2048 * 64;
  const bf16* Vb = V + (size_t)bh * 2048 * 64;

  // ---- prologue: stage Q tile (and, on the async path, K/V chunk 0)
#if ASYNC_LDS
#pragma unroll
  for (int j = 0; j < 4; ++j) {
    int u = j * 256 + tid, row = u >> 3, seg = u & 7;
    async_copy16(Qb + (size_t)(m0 + row) * 64 + seg * 8, sQ + row * 64 + seg * 8);
    async_copy16(Kb + (size_t)row * 64 + seg * 8, smem + row * 64 + seg * 8);
    async_copy16(Vb + (size_t)row * 64 + seg * 8, smem + 16384 + row * 64 + seg * 8);
  }
  async_wait();
#else
#pragma unroll
  for (int j = 0; j < 4; ++j) {
    int u = j * 256 + tid, row = u >> 3, seg = u & 7;
    ((uint4*)sQ)[u] = *(const uint4*)(Qb + (size_t)(m0 + row) * 64 + seg * 8);
  }
#endif
  __syncthreads();
  v16bf aQ[2];
#pragma unroll
  for (int kk = 0; kk < 2; ++kk)
    aQ[kk] = load_frag(sQ + w * 16 * 64, l15, half, 64, 1, kk * 32);

  float m_run[8], l_run[8];
  v8f o[4];
#pragma unroll
  for (int r = 0; r < 8; ++r) { m_run[r] = -3.0e38f; l_run[r] = 0.0f; }
#pragma unroll
  for (int jo = 0; jo < 4; ++jo) o[jo] = zero8();

  int buf = 0;
  for (int kc = 0; kc < 16; ++kc) {
#if ASYNC_LDS
    if (kc + 1 < 16) {                       // overlap next K/V chunk
      int key0 = (kc + 1) * 128;
      bf16* nK = smem + (buf ^ 1) * 8192;
      bf16* nV = smem + 16384 + (buf ^ 1) * 8192;
#pragma unroll
      for (int j = 0; j < 4; ++j) {
        int u = j * 256 + tid, row = u >> 3, seg = u & 7;
        async_copy16(Kb + (size_t)(key0 + row) * 64 + seg * 8,
                     nK + row * 64 + seg * 8);
        async_copy16(Vb + (size_t)(key0 + row) * 64 + seg * 8,
                     nV + row * 64 + seg * 8);
      }
    }
#else
    {
      int key0 = kc * 128;
      __syncthreads();
#pragma unroll
      for (int j = 0; j < 4; ++j) {
        int u = j * 256 + tid, row = u >> 3, seg = u & 7;
        ((uint4*)smem)[u] = *(const uint4*)(Kb + (size_t)(key0 + row) * 64 + seg * 8);
        ((uint4*)(smem + 16384))[u] =
            *(const uint4*)(Vb + (size_t)(key0 + row) * 64 + seg * 8);
      }
      __syncthreads();
    }
#endif
    const bf16* cK = smem + buf * 8192;
    const bf16* cV = smem + 16384 + buf * 8192;

    // S[16x128] = Q . K^T
    v8f s[8];
#pragma unroll
    for (int j = 0; j < 8; ++j) {
      s[j] = zero8();
#pragma unroll
      for (int kk = 0; kk < 2; ++kk) {
        v16bf bK = load_frag(cK + j * 16 * 64, l15, half, 64, 1, kk * 32);
        s[j] = wmma_bf16(aQ[kk], bK, s[j]);
      }
    }

    // Online softmax (rows r+8*half; columns across lanes of each half)
    float alpha[8], rs[8];
#pragma unroll
    for (int r = 0; r < 8; ++r) {
      float mx = s[0][r];
#pragma unroll
      for (int j = 1; j < 8; ++j) mx = fmaxf(mx, s[j][r]);
#pragma unroll
      for (int off = 1; off < 16; off <<= 1)
        mx = fmaxf(mx, __shfl_xor(mx, off, 32));
      float mnew = fmaxf(m_run[r], mx);
      alpha[r] = __expf(m_run[r] - mnew);
      m_run[r] = mnew;
      rs[r] = 0.0f;
    }
#pragma unroll
    for (int j = 0; j < 8; ++j)
#pragma unroll
      for (int r = 0; r < 8; ++r) {
        float p = __expf(s[j][r] - m_run[r]);
        rs[r] += p;
        sP[w * 2048 + (r + 8 * half) * 128 + j * 16 + l15] = (bf16)p;
      }
#pragma unroll
    for (int r = 0; r < 8; ++r) {
#pragma unroll
      for (int off = 1; off < 16; off <<= 1) rs[r] += __shfl_xor(rs[r], off, 32);
      l_run[r] = l_run[r] * alpha[r] + rs[r];
    }
#pragma unroll
    for (int jo = 0; jo < 4; ++jo)
#pragma unroll
      for (int r = 0; r < 8; ++r) o[jo][r] *= alpha[r];
    __syncthreads();                          // order sP writes vs reads

    // O[16x64] += P[16x128] . V[128x64]
    v16bf aP[4];
#pragma unroll
    for (int kk = 0; kk < 4; ++kk)
      aP[kk] = load_frag(sP + w * 2048, l15, half, 128, 1, kk * 32);
#pragma unroll
    for (int jo = 0; jo < 4; ++jo)
#pragma unroll
      for (int kk = 0; kk < 4; ++kk) {
        v16bf bV = load_frag(cV + jo * 16, l15, half, 1, 64, kk * 32);
        o[jo] = wmma_bf16(aP[kk], bV, o[jo]);
      }
#if ASYNC_LDS
    async_wait();
    __syncthreads();
    buf ^= 1;
#endif
  }

  // Normalize and scatter to [B, N, C] bf16 (C index = h*64 + d).
  const int b = bh >> 4, h = bh & 15;
#pragma unroll
  for (int jo = 0; jo < 4; ++jo)
#pragma unroll
    for (int r = 0; r < 8; ++r) {
      int i = m0 + w * 16 + r + 8 * half;
      int d = jo * 16 + l15;
      float vv = o[jo][r] / l_run[r];
      O[(size_t)(b * 2048 + i) * 1024 + h * 64 + d] = (bf16)vv;
    }
}

// ---------------- launcher --------------------------------------------------

extern "C" void kernel_launch(void* const* d_in, const int* in_sizes, int n_in,
                              void* d_out, int out_size, void* d_ws, size_t ws_size,
                              hipStream_t stream) {
  const float* x      = (const float*)d_in[0];  // [2,2048,1024]
  const float* qkv_w  = (const float*)d_in[1];  // [3072,1024]
  const float* qkv_b  = (const float*)d_in[2];  // [3072]
  const float* proj_w = (const float*)d_in[3];  // [1024,1024]
  const float* proj_b = (const float*)d_in[4];  // [1024]
  float* out = (float*)d_out;

  bf16* ws    = (bf16*)d_ws;
  bf16* xb    = ws;               // 4,194,304
  bf16* wqkv  = ws + 4194304;     // 3,145,728
  bf16* wproj = ws + 7340032;     // 1,048,576
  bf16* qkv   = ws + 8388608;     // 3 x 4,194,304 (q,k,v as [B,H,N,Dh])
  bf16* attn  = ws + 20971520;    // 4,194,304 [B,N,C]
  (void)in_sizes; (void)n_in; (void)out_size; (void)ws_size;

  cvt_bf16<<<2048, 256, 0, stream>>>(x, xb, 4194304);
  cvt_bf16<<<2048, 256, 0, stream>>>(qkv_w, wqkv, 3145728);
  cvt_bf16<<<1024, 256, 0, stream>>>(proj_w, wproj, 1048576);

  gemm_bf16_kernel<256, 4, 0><<<dim3(12, 32), 256, 0, stream>>>(xb, wqkv, qkv_b, qkv);
  attn_fwd<<<dim3(16, 32), 256, 0, stream>>>(qkv, qkv + 4194304, qkv + 2 * 4194304,
                                             attn);
  gemm_bf16_kernel<128, 2, 1><<<dim3(8, 32), 256, 0, stream>>>(attn, wproj, proj_b,
                                                               out);
}
